// RNNEncoder_11630771438225
// MI455X (gfx1250) — compile-verified, matrix operand order checked by reading, not run on hardware
//
#include <hip/hip_runtime.h>

// ---------------------------------------------------------------------------
// AWD-LSTM RNNEncoder inference for MI455X (gfx1250, wave32, WMMA).
//   emb lookup -> 3x (chunked bf16 WMMA GEMM for x-projection with
//   fragment-layout XG, then per-timestep fused recurrent WMMA + LSTM cell).
// Gate/weight column space is per-quadrant padded: n' = q*Jpad + j,
// Jpad = ceil16(H), so every 16x16 tile is quadrant-pure and XG can live in
// WMMA D-fragment layout (contiguous 32B per lane per tile).
// ---------------------------------------------------------------------------

typedef __attribute__((ext_vector_type(16))) __bf16 v16bf;
typedef __attribute__((ext_vector_type(8)))  __bf16 v8bf;
typedef __attribute__((ext_vector_type(8)))  float  v8f;

#define SL   512
#define BS   64
#define EMB  400
#define CH   64          // timesteps per XG chunk
#define CHROWS (CH * BS) // 4096 rows per chunk

__device__ __forceinline__ __bf16 f2bf(float f) { return (__bf16)f; }

// Load one 16x32 bf16 WMMA fragment (A: row=M, or B: row=N; both row-major
// [row, K]).  Lane L (L&15 = row) holds K = kb..kb+7 in elems 0..7 and
// K = kb+16..kb+23 in elems 8..15, kb = kk + 8*(L>=16): two b128 loads.
__device__ __forceinline__ v16bf load_frag(const __bf16* __restrict__ base,
                                           int stride, int row0, int kk, int lane) {
    int r  = row0 + (lane & 15);
    int kb = kk + ((lane >> 4) << 3);
    const __bf16* p = base + (size_t)r * (size_t)stride + kb;
    v8bf lo = *(const v8bf*)(p);
    v8bf hi = *(const v8bf*)(p + 16);
    v16bf out;
#pragma unroll
    for (int i = 0; i < 8; ++i) { out[i] = lo[i]; out[i + 8] = hi[i]; }
    return out;
}

// ------------------------------- utility kernels ---------------------------

__global__ void zero_bf16_kernel(__bf16* p, size_t n) {
    size_t i = (size_t)blockIdx.x * blockDim.x + threadIdx.x;
    if (i < n) p[i] = f2bf(0.0f);
}

__global__ void zero_f32_kernel(float* p, size_t n) {
    size_t i = (size_t)blockIdx.x * blockDim.x + threadIdx.x;
    if (i < n) p[i] = 0.0f;
}

// Convert a [4H, cols] f32 weight into quadrant-padded bf16 [4*Jpad, Cpad]:
// dst row n' = q*Jpad + j  <-  src row q*H + j (zeros for j>=H or k>=cols).
__global__ void conv_quad_bf16_kernel(const float* __restrict__ src, __bf16* __restrict__ dst,
                                      int H, int cols, int Jpad, int Cpad) {
    size_t i = (size_t)blockIdx.x * blockDim.x + threadIdx.x;
    size_t total = (size_t)4 * Jpad * Cpad;
    if (i >= total) return;
    int n = (int)(i / Cpad);
    int k = (int)(i % Cpad);
    int q = n / Jpad;
    int j = n % Jpad;
    float v = (j < H && k < cols) ? src[(size_t)(q * H + j) * cols + k] : 0.0f;
    dst[i] = f2bf(v);
}

// Combined bias in quadrant-padded layout: out[q*Jpad + j] = b0+b1 at q*H+j.
__global__ void bias_combine_kernel(const float* __restrict__ b0, const float* __restrict__ b1,
                                    float* __restrict__ out, int H, int Jpad) {
    int i = blockIdx.x * blockDim.x + threadIdx.x;
    if (i >= 4 * Jpad) return;
    int q = i / Jpad;
    int j = i % Jpad;
    out[i] = (j < H) ? (b0[q * H + j] + b1[q * H + j]) : 0.0f;
}

// Embedding gather -> bf16 layer-0 input, row stride Kpad0 (pads pre-zeroed).
__global__ void embed_kernel(const int* __restrict__ tokens, const float* __restrict__ emb,
                             __bf16* __restrict__ X, int Kpad0) {
    size_t i = (size_t)blockIdx.x * blockDim.x + threadIdx.x;
    size_t total = (size_t)SL * BS * EMB;
    if (i >= total) return;
    int r = (int)(i / EMB);
    int k = (int)(i % EMB);
    int tok = tokens[r];
    X[(size_t)r * Kpad0 + k] = f2bf(emb[(size_t)tok * EMB + k]);
}

// --------------------- input-projection GEMM (per chunk) -------------------
// 2x2 register-blocked: each wave owns a 32x32 macro-tile (2 m-tiles x
// 2 n-tiles) -> 4 WMMAs per 4 fragment loads (A reused across n, B across m).
// XGf stored in D-fragment layout: tile (rowTile, colTile) occupies 256
// contiguous floats at ((rowTile*NT + colTile) << 8), lane*8 per lane.
// grid: (CHROWS/128, NPq/32), block: 128 (4 waves).
__global__ void gemm_xg_kernel(const __bf16* __restrict__ X, const __bf16* __restrict__ W,
                               const float* __restrict__ bias, float* __restrict__ XGf,
                               int Kpad, int NT, int rowBase) {
    int lane = threadIdx.x & 31;
    int wave = threadIdx.x >> 5;
    int mLocal = blockIdx.x * 128 + wave * 32;   // 2 m-tiles: mLocal, mLocal+16
    int n0 = blockIdx.y * 32;                    // 2 n-tiles: n0, n0+16

    v8f acc[2][2];
#pragma unroll
    for (int ni = 0; ni < 2; ++ni) {
        float bv = bias[n0 + ni * 16 + (lane & 15)];
#pragma unroll
        for (int mi = 0; mi < 2; ++mi)
#pragma unroll
            for (int v = 0; v < 8; ++v) acc[mi][ni][v] = bv;
    }

    for (int kk = 0; kk < Kpad; kk += 32) {
        v16bf a0 = load_frag(X, Kpad, rowBase + mLocal,      kk, lane);
        v16bf a1 = load_frag(X, Kpad, rowBase + mLocal + 16, kk, lane);
        v16bf b0 = load_frag(W, Kpad, n0,      kk, lane);
        v16bf b1 = load_frag(W, Kpad, n0 + 16, kk, lane);
        acc[0][0] = __builtin_amdgcn_wmma_f32_16x16x32_bf16(false, a0, false, b0,
                                                            (short)0, acc[0][0], false, false);
        acc[0][1] = __builtin_amdgcn_wmma_f32_16x16x32_bf16(false, a0, false, b1,
                                                            (short)0, acc[0][1], false, false);
        acc[1][0] = __builtin_amdgcn_wmma_f32_16x16x32_bf16(false, a1, false, b0,
                                                            (short)0, acc[1][0], false, false);
        acc[1][1] = __builtin_amdgcn_wmma_f32_16x16x32_bf16(false, a1, false, b1,
                                                            (short)0, acc[1][1], false, false);
    }

    int rowTile0 = blockIdx.x * 8 + wave * 2;
    int colTile0 = blockIdx.y * 2;
#pragma unroll
    for (int mi = 0; mi < 2; ++mi)
#pragma unroll
        for (int ni = 0; ni < 2; ++ni) {
            size_t base = (((size_t)(rowTile0 + mi) * NT + (colTile0 + ni)) << 8)
                        + (size_t)lane * 8;
            *(v8f*)(XGf + base) = acc[mi][ni];   // contiguous 32B store per lane
        }
}

// ------------------ fused recurrent step (one launch per t) ----------------
// Each block owns hidden cols [j0, j0+16); its 4 waves own 16 batch rows each
// and accumulate the four gate quadrants i,f,g,o with WMMA over K = H.
// XG tiles are read back in fragment layout (one v8f per quadrant) and the
// LSTM cell update is fully lane-local.
__global__ void lstm_step_kernel(const __bf16* __restrict__ Whh,
                                 const float*  __restrict__ XGf,
                                 const __bf16* __restrict__ hprev,
                                 __bf16*       __restrict__ hnext,
                                 float*        __restrict__ c,
                                 float*        __restrict__ outh,
                                 __bf16*       __restrict__ Xnext,
                                 int H, int Hpad, int Jpad, int NT,
                                 int tLocal, int t, int KnextPad) {
    int lane = threadIdx.x & 31;
    int wave = threadIdx.x >> 5;
    int m0 = wave * 16;           // batch tile
    int j0 = blockIdx.x * 16;     // hidden-unit tile

    v8f acc[4];
    {   // init accumulators from fragment-layout XG (coalesced v8f loads)
        int rowTile = tLocal * 4 + wave;  // (tLocal*BS + m0) / 16
        int jt = Jpad >> 4;
#pragma unroll
        for (int q = 0; q < 4; ++q) {
            int colTile = q * jt + blockIdx.x;
            size_t base = (((size_t)rowTile * NT + colTile) << 8) + (size_t)lane * 8;
            acc[q] = *(const v8f*)(XGf + base);
        }
    }

    for (int kk = 0; kk < Hpad; kk += 32) {
        v16bf a = load_frag(hprev, Hpad, m0, kk, lane);
#pragma unroll
        for (int q = 0; q < 4; ++q) {
            v16bf b = load_frag(Whh, Hpad, q * Jpad + j0, kk, lane);
            acc[q] = __builtin_amdgcn_wmma_f32_16x16x32_bf16(false, a, false, b,
                                                             (short)0, acc[q], false, false);
        }
    }

    int j  = j0 + (lane & 15);
    int mb = m0 + ((lane >> 4) << 3);
    if (j < H) {
#pragma unroll
        for (int v = 0; v < 8; ++v) {
            int m = mb + v;
            float si = 1.0f / (1.0f + __expf(-acc[0][v]));
            float sf = 1.0f / (1.0f + __expf(-acc[1][v]));
            float tg = tanhf(acc[2][v]);
            float so = 1.0f / (1.0f + __expf(-acc[3][v]));
            size_t ci = (size_t)m * Hpad + j;
            float cn = sf * c[ci] + si * tg;
            c[ci] = cn;
            float h = so * tanhf(cn);
            outh[((size_t)t * BS + m) * H + j] = h;
            hnext[ci] = f2bf(h);
            if (Xnext) Xnext[((size_t)t * BS + m) * KnextPad + j] = f2bf(h);
        }
    }
}

// --------------------------------- launch ----------------------------------

static inline size_t align256(size_t x) { return (x + 255) & ~(size_t)255; }

extern "C" void kernel_launch(void* const* d_in, const int* in_sizes, int n_in,
                              void* d_out, int out_size, void* d_ws, size_t ws_size,
                              hipStream_t stream) {
    (void)in_sizes; (void)n_in; (void)out_size; (void)ws_size;

    const int*   tokens = (const int*)d_in[0];
    const float* embw   = (const float*)d_in[1];
    const float* wih[3], *whh[3], *bih[3], *bhh[3];
    for (int l = 0; l < 3; ++l) {
        wih[l] = (const float*)d_in[2 + 4 * l];
        whh[l] = (const float*)d_in[3 + 4 * l];
        bih[l] = (const float*)d_in[4 + 4 * l];
        bhh[l] = (const float*)d_in[5 + 4 * l];
    }
    float* out = (float*)d_out;

    const int Iin[3]  = {400, 1150, 1150};
    const int Hout[3] = {1150, 1150, 400};
    int Kpad[3], Hpad[3], Jpad[3], NPq[3];
    for (int l = 0; l < 3; ++l) {
        Kpad[l] = ((Iin[l] + 31) / 32) * 32;      // 416, 1152, 1152
        Hpad[l] = ((Hout[l] + 31) / 32) * 32;     // 1152, 1152, 416
        Jpad[l] = ((Hout[l] + 15) / 16) * 16;     // 1152, 1152, 400
        NPq[l]  = 4 * Jpad[l];                    // 4608, 4608, 1600
    }
    size_t outBase[3];
    outBase[0] = 0;
    outBase[1] = (size_t)SL * BS * Hout[0];
    outBase[2] = outBase[1] + (size_t)SL * BS * Hout[1];

    // ---- workspace carve-up (all offsets 256B aligned) ----
    const size_t XBF_ELEMS = (size_t)SL * BS * 1152;   // max padded X
    char* w = (char*)d_ws;
    __bf16* XbfA  = (__bf16*)w; w += align256(XBF_ELEMS * sizeof(__bf16));
    __bf16* XbfB  = (__bf16*)w; w += align256(XBF_ELEMS * sizeof(__bf16));
    float*  XGf   = (float*)w;  w += align256((size_t)CHROWS * 4608 * sizeof(float));
    __bf16* wihbf = (__bf16*)w; w += align256((size_t)4608 * 1152 * sizeof(__bf16));
    __bf16* whhbf = (__bf16*)w; w += align256((size_t)4608 * 1152 * sizeof(__bf16));
    float*  biasb = (float*)w;  w += align256((size_t)4608 * sizeof(float));
    __bf16* hA    = (__bf16*)w; w += align256((size_t)BS * 1152 * sizeof(__bf16));
    __bf16* hB    = (__bf16*)w; w += align256((size_t)BS * 1152 * sizeof(__bf16));
    float*  cbuf  = (float*)w;  w += align256((size_t)BS * 1152 * sizeof(float));

    __bf16* Xin  = XbfA;
    __bf16* Xout = XbfB;

    // ---- layer-0 input: zero pads then embedding gather ----
    {
        size_t n = XBF_ELEMS;
        zero_bf16_kernel<<<(unsigned)((n + 255) / 256), 256, 0, stream>>>(Xin, n);
        size_t tot = (size_t)SL * BS * EMB;
        embed_kernel<<<(unsigned)((tot + 255) / 256), 256, 0, stream>>>(tokens, embw, Xin, Kpad[0]);
    }

    for (int l = 0; l < 3; ++l) {
        const int H = Hout[l], I = Iin[l];
        const int kp = Kpad[l], hp = Hpad[l], jp = Jpad[l], np = NPq[l];
        const int NT = np / 16;

        // weight / bias conversion to quadrant-padded bf16
        {
            size_t tot = (size_t)np * kp;
            conv_quad_bf16_kernel<<<(unsigned)((tot + 255) / 256), 256, 0, stream>>>(
                wih[l], wihbf, H, I, jp, kp);
            tot = (size_t)np * hp;
            conv_quad_bf16_kernel<<<(unsigned)((tot + 255) / 256), 256, 0, stream>>>(
                whh[l], whhbf, H, H, jp, hp);
            bias_combine_kernel<<<(np + 255) / 256, 256, 0, stream>>>(bih[l], bhh[l], biasb, H, jp);
        }
        // zero recurrent state (h ping-pong + c)
        {
            size_t n = (size_t)BS * hp;
            zero_bf16_kernel<<<(unsigned)((n + 255) / 256), 256, 0, stream>>>(hA, n);
            zero_bf16_kernel<<<(unsigned)((n + 255) / 256), 256, 0, stream>>>(hB, n);
            zero_f32_kernel<<<(unsigned)((n + 255) / 256), 256, 0, stream>>>(cbuf, n);
        }
        // zero next layer's input buffer (pads must be 0)
        if (l < 2) {
            size_t n = XBF_ELEMS;
            zero_bf16_kernel<<<(unsigned)((n + 255) / 256), 256, 0, stream>>>(Xout, n);
        }

        __bf16* hp0 = hA;
        __bf16* hp1 = hB;
        const int jt = (H + 15) / 16;
        const int knext = ((H + 31) / 32) * 32;   // next layer's padded input stride

        for (int chunk = 0; chunk < SL / CH; ++chunk) {
            // XGf = frag(Xin @ Wih^T + bias) for this chunk of timesteps
            dim3 ggrid(CHROWS / 128, np / 32);
            gemm_xg_kernel<<<ggrid, 128, 0, stream>>>(Xin, wihbf, biasb, XGf,
                                                      kp, NT, chunk * CHROWS);
            for (int tl = 0; tl < CH; ++tl) {
                int t = chunk * CH + tl;
                __bf16* xnext = (l < 2) ? Xout : (__bf16*)nullptr;
                lstm_step_kernel<<<jt, 128, 0, stream>>>(
                    whhbf, XGf, hp0, hp1, cbuf, out + outBase[l], xnext,
                    H, hp, jp, NT, tl, t, (l < 2) ? knext : 0);
                __bf16* tmp = hp0; hp0 = hp1; hp1 = tmp;
            }
        }

        // next layer consumes this layer's output
        __bf16* tmp = Xin; Xin = Xout; Xout = tmp;
    }
}
